// LinearRNN_3341484556687
// MI455X (gfx1250) — matrix operand order repaired and checked by reading, not compiled, not run on hardware
//
#include <hip/hip_runtime.h>

#define B_    16
#define T_    4096
#define DIN_  256
#define H_    256
#define LCH   32            // scan chunk length (rows per GEMM tile block)
#define NCH   (T_ / LCH)    // 128 chunks
#define SA_STRIDE 264       // bf16 elems per padded LDS row (conflict-free)
#define SU_STRIDE 260       // f32 elems per padded LDS row

typedef __attribute__((ext_vector_type(16))) __bf16 v16bf;
typedef __attribute__((ext_vector_type(8)))  float  v8f;

union FragBF { uint4 q[2]; v16bf v; };

__device__ __forceinline__ unsigned short f2bf(float f) {
  unsigned int u = __float_as_uint(f);
  u += 0x7FFFu + ((u >> 16) & 1u);   // round-to-nearest-even
  return (unsigned short)(u >> 16);
}

// ---------------- kernel 0: weight prep (fp32 -> bf16, extract diag) --------
__global__ void lrnn_prep(const float* __restrict__ Wi, const float* __restrict__ Wh,
                          const float* __restrict__ Wo,
                          unsigned short* __restrict__ Wi_bf,
                          unsigned short* __restrict__ Wo_bf,
                          float* __restrict__ a) {
  int idx = blockIdx.x * blockDim.x + threadIdx.x;
  if (idx < H_ * DIN_) {
    Wi_bf[idx] = f2bf(Wi[idx]);
    Wo_bf[idx] = f2bf(Wo[idx]);
  }
  if (idx < H_) a[idx] = Wh[idx * H_ + idx];   // diagonal recurrence gain
}

// ---------------- kernel 1: u = x@Wi^T + bi  (WMMA)  + chunk-local scan -----
__global__ __launch_bounds__(256) void lrnn_gemm_scan(
    const float* __restrict__ x, const unsigned short* __restrict__ Wi_bf,
    const float* __restrict__ bi, const float* __restrict__ bh,
    const float* __restrict__ a,
    float* __restrict__ y /* = d_out, chunk-local scan output */,
    float* __restrict__ y_end) {
  __shared__ unsigned short sA[LCH * SA_STRIDE];   // x chunk, bf16
  __shared__ float          sU[LCH * SU_STRIDE];   // u chunk, f32

  const int tid = threadIdx.x;
  const int b   = blockIdx.x / NCH;
  const int c   = blockIdx.x % NCH;
  const int t0  = c * LCH;

  // phase 1: stream x chunk -> bf16 in LDS
  const float4* xg = (const float4*)(x + ((size_t)b * T_ + t0) * DIN_);
  #pragma unroll
  for (int i = 0; i < (LCH * DIN_ / 4) / 256; ++i) {
    int f   = tid + i * 256;      // float4 index
    int row = f >> 6;             // 64 float4 per row
    int c4  = f & 63;
    float4 v = xg[row * 64 + c4];
    uint2 pk;
    pk.x = (unsigned)f2bf(v.x) | ((unsigned)f2bf(v.y) << 16);
    pk.y = (unsigned)f2bf(v.z) | ((unsigned)f2bf(v.w) << 16);
    *(uint2*)&sA[row * SA_STRIDE + c4 * 4] = pk;
  }
  __syncthreads();

  // phase 2: WMMA bf16 / f32 acc. Each wave owns 2 t-tiles x 2 h-tiles:
  // every B fragment loaded exactly once per workgroup, A reused from LDS.
  const int lane   = tid & 31;
  const int wave   = tid >> 5;        // h-tile pair {2w, 2w+1}
  const int ln16   = lane & 15;
  const int hiHalf = lane >> 4;       // 0 or 1

  v8f acc[2][2] = {};
  #pragma unroll
  for (int kb = 0; kb < DIN_; kb += 32) {
    FragBF a0, a1;
    const unsigned short* ap0 = &sA[(ln16)      * SA_STRIDE + kb + hiHalf * 8];
    const unsigned short* ap1 = &sA[(16 + ln16) * SA_STRIDE + kb + hiHalf * 8];
    a0.q[0] = *(const uint4*)(ap0);
    a0.q[1] = *(const uint4*)(ap0 + 16);
    a1.q[0] = *(const uint4*)(ap1);
    a1.q[1] = *(const uint4*)(ap1 + 16);
    #pragma unroll
    for (int n = 0; n < 2; ++n) {
      FragBF bfr;
      const unsigned short* bp =
          Wi_bf + (size_t)((wave * 2 + n) * 16 + ln16) * DIN_ + kb + hiHalf * 8;
      bfr.q[0] = *(const uint4*)(bp);
      bfr.q[1] = *(const uint4*)(bp + 16);
      acc[0][n] = __builtin_amdgcn_wmma_f32_16x16x32_bf16(
          false, a0.v, false, bfr.v, (short)0, acc[0][n], false, false);
      acc[1][n] = __builtin_amdgcn_wmma_f32_16x16x32_bf16(
          false, a1.v, false, bfr.v, (short)0, acc[1][n], false, false);
    }
  }

  // phase 3: u + bi -> sU (C/D layout: elem e -> row e(+8), col lane%16)
  #pragma unroll
  for (int t = 0; t < 2; ++t) {
    #pragma unroll
    for (int n = 0; n < 2; ++n) {
      int col = (wave * 2 + n) * 16 + ln16;
      float bic = bi[col];
      #pragma unroll
      for (int e = 0; e < 8; ++e) {
        int row = t * 16 + e + hiHalf * 8;
        sU[row * SU_STRIDE + col] = acc[t][n][e] + bic;
      }
    }
  }
  __syncthreads();

  // phase 4: chunk-local linear scan, one channel per thread, zero init
  {
    int j = tid;
    float aj = a[j], bhj = bh[j], h = 0.0f;
    float* yrow = y + ((size_t)b * T_ + t0) * H_ + j;
    #pragma unroll 4
    for (int l = 0; l < LCH; ++l) {
      h = fmaf(aj, h, sU[l * SU_STRIDE + j] + bhj);
      yrow[(size_t)l * H_] = h;
    }
    y_end[((size_t)b * NCH + c) * H_ + j] = h;
  }
}

// ---------------- kernel 2: sequential carry combine + power table ----------
__global__ void lrnn_carry(const float* __restrict__ a, const float* __restrict__ y_end,
                           float* __restrict__ carry_in, float* __restrict__ apow) {
  int idx = blockIdx.x * blockDim.x + threadIdx.x;   // B_*H_ = 4096 threads
  int b = idx >> 8, j = idx & 255;
  float aj = a[j];
  if (idx < H_) {           // a^(l+1) table, l = 0..LCH-1
    float p = aj;
    for (int l = 0; l < LCH; ++l) { apow[l * H_ + j] = p; p *= aj; }
  }
  float aL = aj;
  #pragma unroll
  for (int i = 0; i < 5; ++i) aL *= aL;   // a^32
  float carry = 0.0f;
  for (int cc = 0; cc < NCH; ++cc) {
    size_t o = ((size_t)b * NCH + cc) * H_ + j;
    carry_in[o] = carry;                       // hidden state entering chunk cc
    carry = y_end[o] + aL * carry;
  }
}

// ---------------- kernel 3: fixup hidden = y + a^{l+1}*carry, out GEMM ------
__global__ __launch_bounds__(256) void lrnn_fix_gemm_out(
    const unsigned short* __restrict__ Wo_bf, const float* __restrict__ bo,
    const float* __restrict__ carry_in, const float* __restrict__ apow,
    float* __restrict__ out /* holds y, overwritten in place per block */) {
  __shared__ unsigned short sA[LCH * SA_STRIDE];

  const int tid = threadIdx.x;
  const int b   = blockIdx.x / NCH;
  const int c   = blockIdx.x % NCH;
  const int t0  = c * LCH;
  const float* carr = carry_in + ((size_t)b * NCH + c) * H_;

  // phase 1: read y chunk, apply carry fixup, bf16 -> LDS
  {
    int j = tid;
    float cj = carr[j];
    #pragma unroll
    for (int l = 0; l < LCH; ++l) {
      float yv  = out[((size_t)b * T_ + t0 + l) * H_ + j];
      float hid = fmaf(apow[l * H_ + j], cj, yv);
      sA[l * SA_STRIDE + j] = f2bf(hid);
    }
  }
  __syncthreads();   // all global y reads done before this block overwrites out

  const int lane   = tid & 31;
  const int wave   = tid >> 5;
  const int ln16   = lane & 15;
  const int hiHalf = lane >> 4;

  v8f acc[2][2] = {};
  #pragma unroll
  for (int kb = 0; kb < H_; kb += 32) {
    FragBF a0, a1;
    const unsigned short* ap0 = &sA[(ln16)      * SA_STRIDE + kb + hiHalf * 8];
    const unsigned short* ap1 = &sA[(16 + ln16) * SA_STRIDE + kb + hiHalf * 8];
    a0.q[0] = *(const uint4*)(ap0);
    a0.q[1] = *(const uint4*)(ap0 + 16);
    a1.q[0] = *(const uint4*)(ap1);
    a1.q[1] = *(const uint4*)(ap1 + 16);
    #pragma unroll
    for (int n = 0; n < 2; ++n) {
      FragBF bfr;
      const unsigned short* bp =
          Wo_bf + (size_t)((wave * 2 + n) * 16 + ln16) * H_ + kb + hiHalf * 8;
      bfr.q[0] = *(const uint4*)(bp);
      bfr.q[1] = *(const uint4*)(bp + 16);
      acc[0][n] = __builtin_amdgcn_wmma_f32_16x16x32_bf16(
          false, a0.v, false, bfr.v, (short)0, acc[0][n], false, false);
      acc[1][n] = __builtin_amdgcn_wmma_f32_16x16x32_bf16(
          false, a1.v, false, bfr.v, (short)0, acc[1][n], false, false);
    }
  }

  #pragma unroll
  for (int t = 0; t < 2; ++t) {
    #pragma unroll
    for (int n = 0; n < 2; ++n) {
      int col = (wave * 2 + n) * 16 + ln16;
      float boc = bo[col];
      #pragma unroll
      for (int e = 0; e < 8; ++e) {
        int row = t * 16 + e + hiHalf * 8;
        out[((size_t)b * T_ + t0 + row) * H_ + col] = acc[t][n][e] + boc;
      }
    }
  }
}

// ---------------------------------------------------------------------------
extern "C" void kernel_launch(void* const* d_in, const int* in_sizes, int n_in,
                              void* d_out, int out_size, void* d_ws, size_t ws_size,
                              hipStream_t stream) {
  const float* x  = (const float*)d_in[0];
  const float* Wi = (const float*)d_in[1];
  const float* bi = (const float*)d_in[2];
  const float* Wh = (const float*)d_in[3];
  const float* bh = (const float*)d_in[4];
  const float* Wo = (const float*)d_in[5];
  const float* bo = (const float*)d_in[6];
  float* outp = (float*)d_out;

  // workspace layout (~4.5 MB)
  char* w = (char*)d_ws;
  unsigned short* Wi_bf = (unsigned short*)(w);                  // 128 KiB
  unsigned short* Wo_bf = (unsigned short*)(w + 131072);         // 128 KiB
  float* a       = (float*)(w + 262144);                         // 1 KiB
  float* y_end   = (float*)(w + 263168);                         // B*NCH*H f32 = 2 MiB
  float* carry   = (float*)(w + 263168 + (size_t)B_ * NCH * H_ * 4);
  float* apow    = (float*)(w + 263168 + 2 * (size_t)B_ * NCH * H_ * 4);

  lrnn_prep<<<(H_ * DIN_ + 255) / 256, 256, 0, stream>>>(Wi, Wh, Wo, Wi_bf, Wo_bf, a);
  lrnn_gemm_scan<<<B_ * NCH, 256, 0, stream>>>(x, Wi_bf, bi, bh, a, outp, y_end);
  lrnn_carry<<<(B_ * H_) / 256, 256, 0, stream>>>(a, y_end, carry, apow);
  lrnn_fix_gemm_out<<<B_ * NCH, 256, 0, stream>>>(Wo_bf, bo, carry, apow, outp);
}
